// MaskedMultiHeadAttention_42966852829769
// MI455X (gfx1250) — matrix operand order repaired and checked by reading, not compiled, not run on hardware
//
#include <hip/hip_runtime.h>
#include <hip/hip_bf16.h>

#define B_ 2
#define S_ 2048
#define D_ 1024
#define H_ 16
#define E_ 64
#define BH_ (B_*H_)
#define BS_ (B_*S_)

typedef __attribute__((ext_vector_type(16))) __bf16 bf16x16;
typedef __attribute__((ext_vector_type(2)))  __bf16 bf16x2;
typedef __attribute__((ext_vector_type(8)))  float  f32x8;

union ABreg { bf16x16 v; unsigned u[8]; };

#if defined(__has_builtin)
#if __has_builtin(__builtin_amdgcn_global_load_async_to_lds_b128) && \
    __has_builtin(__builtin_amdgcn_s_wait_asynccnt)
#define USE_ASYNC_LDS 1
#endif
#endif

#ifdef USE_ASYNC_LDS
typedef int v4i __attribute__((vector_size(16)));
typedef __attribute__((address_space(1))) v4i* as1_v4i_p;
typedef __attribute__((address_space(3))) v4i* as3_v4i_p;
#endif

// Hardware bf16 converts (backend picks v_cvt_pk_bf16_f32-class ops)
__device__ __forceinline__ unsigned short f2bf(float f) {
  __bf16 h = (__bf16)f;
  return __builtin_bit_cast(unsigned short, h);
}
__device__ __forceinline__ unsigned packbf(float lo, float hi) {
  bf16x2 p;
  p.x = (__bf16)lo;
  p.y = (__bf16)hi;
  return __builtin_bit_cast(unsigned, p);
}

// ---- WMMA operand loaders (v_wmma_f32_16x16x32_bf16 layouts, wave32) -------
// A tile 16(M) x 32(K), source bf16 row-major, stride ld.
// Lane: M = lane&15; VGPR i<4 -> K=2i(+8 if lane>=16); i>=4 -> 16+2(i-4)(+8 hi)
__device__ __forceinline__ void loadA_bf(const unsigned short* p, int ld, ABreg& a) {
  int lane = threadIdx.x & 31;
  const unsigned short* row = p + (size_t)(lane & 15) * ld + (((lane >> 4) & 1) ? 8 : 0);
#pragma unroll
  for (int i = 0; i < 4; ++i) a.u[i]     = *(const unsigned*)(row + 2 * i);
#pragma unroll
  for (int i = 0; i < 4; ++i) a.u[4 + i] = *(const unsigned*)(row + 16 + 2 * i);
}
// B tile 32(K) x 16(N): source N-major (row n holds K contiguous), stride ld.
// Lane: N = lane&15; VGPR i<4 -> K=2i(+8 if lane>=16); i>=4 -> 16+2(i-4)(+8 hi)
__device__ __forceinline__ void loadB_bf(const unsigned short* p, int ld, ABreg& b) {
  int lane = threadIdx.x & 31;
  const unsigned short* row = p + (size_t)(lane & 15) * ld + (((lane >> 4) & 1) ? 8 : 0);
#pragma unroll
  for (int i = 0; i < 8; ++i) b.u[i] = *(const unsigned*)(row + 2 * i + (i >= 4 ? 8 : 0));
}
// B tile from an LDS slab laid out [64 rows][32 k] bf16
__device__ __forceinline__ void loadB_lds(const unsigned short* lds, int nbase, ABreg& b) {
  int lane = threadIdx.x & 31;
  const unsigned short* row = lds + (nbase + (lane & 15)) * 32 + (((lane >> 4) & 1) ? 8 : 0);
#pragma unroll
  for (int i = 0; i < 8; ++i) b.u[i] = *(const unsigned*)(row + 2 * i + (i >= 4 ? 8 : 0));
}
__device__ __forceinline__ f32x8 wmma_bf16(const ABreg& a, const ABreg& b, f32x8 c) {
  return __builtin_amdgcn_wmma_f32_16x16x32_bf16(false, a.v, false, b.v, (short)0, c, false, false);
}

// Stage a 64x32 bf16 B slab (4KB) into LDS: 256 threads x 16B, async if available
__device__ __forceinline__ void stage_B(const unsigned short* gsrc_row0, int ld,
                                        unsigned short* lds) {
  int tid = threadIdx.x;
  int n  = tid >> 2;         // 0..63 (slab row)
  int kc = (tid & 3) * 8;    // 0,8,16,24 (element offset in row)
  const unsigned short* g = gsrc_row0 + (size_t)n * ld + kc;
  unsigned short* l = lds + n * 32 + kc;
#ifdef USE_ASYNC_LDS
  __builtin_amdgcn_global_load_async_to_lds_b128(
      (as1_v4i_p)(__attribute__((address_space(1))) void*)(void*)g,
      (as3_v4i_p)(__attribute__((address_space(3))) void*)(void*)l,
      0, 0);
#else
  *(uint4*)l = *(const uint4*)g;
#endif
}
__device__ __forceinline__ void stage_wait() {
#ifdef USE_ASYNC_LDS
  __builtin_amdgcn_s_wait_asynccnt(0);
#endif
  __syncthreads();
}

// ---- K0a: fp32 -> bf16 elementwise (vectorized 4-wide) ---------------------
__global__ __launch_bounds__(256) void k_cvt(const float4* __restrict__ src,
                                             uint2* __restrict__ dst) {
  size_t i = (size_t)blockIdx.x * 256 + threadIdx.x;
  float4 f = src[i];
  uint2 o;
  o.x = packbf(f.x, f.y);
  o.y = packbf(f.z, f.w);
  dst[i] = o;
}

// ---- K0b: weight transposes to bf16 ----------------------------------------
__global__ __launch_bounds__(256) void k_transposeW(const float* __restrict__ W,
                                                    unsigned short* __restrict__ WT) {
  int idx = blockIdx.x * 256 + threadIdx.x;            // over H*D*E
  int h = idx / (D_ * E_);
  int r = idx - h * (D_ * E_);
  int d = r / E_;
  int e = r - d * E_;
  WT[((size_t)h * E_ + e) * D_ + d] = f2bf(W[idx]);    // [H,E,D]
}
__global__ __launch_bounds__(256) void k_transposeWo(const float* __restrict__ Wo,
                                                     unsigned short* __restrict__ WoT) {
  int idx = blockIdx.x * 256 + threadIdx.x;            // over D*D
  int d = idx / D_;
  int n = idx - d * D_;
  WoT[(size_t)n * D_ + d] = f2bf(Wo[idx]);             // [N,D]
}

// ---- K1: per-head projections (q/k -> [bh,s,e]; v -> transposed [bh,e,s]) --
// B slab (the head's 64xD weight matrix, 32-K slices) is shared by all 8 waves:
// async-staged to LDS, double buffered.
__global__ __launch_bounds__(256) void k_qkv_proj(const unsigned short* __restrict__ X,  // [B,S,D] bf16
                                                  const unsigned short* __restrict__ WT, // [H,E,D] bf16
                                                  const float* __restrict__ bias,        // [H,E]
                                                  unsigned short* __restrict__ out,
                                                  int transpose_out) {
  __shared__ unsigned short Bs[2][64 * 32];
  int w = threadIdx.x >> 5, lane = threadIdx.x & 31;
  int bh = blockIdx.y, b = bh >> 4, h = bh & 15;
  int r0 = blockIdx.x * 128 + w * 16;
  const unsigned short* Xrow = X + ((size_t)b * S_ + r0) * D_;
  const unsigned short* Wh = WT + (size_t)h * E_ * D_;

  f32x8 acc[4] = {};
  stage_B(Wh, D_, Bs[0]);
  int cur = 0;
  for (int d0 = 0; d0 < D_; d0 += 32) {
    stage_wait();                                   // slab[cur] ready, prior compute done
    if (d0 + 32 < D_) stage_B(Wh + d0 + 32, D_, Bs[cur ^ 1]);   // overlap DMA w/ WMMA
    ABreg a; loadA_bf(Xrow + d0, D_, a);
#pragma unroll
    for (int t = 0; t < 4; ++t) {
      ABreg bb; loadB_lds(Bs[cur], t * 16, bb);
      acc[t] = wmma_bf16(a, bb, acc[t]);
    }
    __syncthreads();                                // done with slab[cur]
    cur ^= 1;
  }

  int n = lane & 15, hi = (lane >> 4) & 1;
#pragma unroll
  for (int t = 0; t < 4; ++t) {
    int col = t * 16 + n;
    float bvv = bias[h * E_ + col];
#pragma unroll
    for (int c = 0; c < 8; ++c) {
      int row = r0 + c + hi * 8;
      float val = acc[t][c] + bvv;
      size_t off = transpose_out ? ((size_t)(bh * E_ + col) * S_ + row)
                                 : ((size_t)((size_t)bh * S_ + row) * E_ + col);
      out[off] = f2bf(val);
    }
  }
}

// ---- K2a: causal raw scores -> weights buffer ------------------------------
__global__ __launch_bounds__(256) void k_scores(const unsigned short* __restrict__ qh,  // [BH,S,E]
                                                const unsigned short* __restrict__ kh,  // [BH,S,E]
                                                float* __restrict__ wts) {              // [BH,S,S]
  int w = threadIdx.x >> 5, lane = threadIdx.x & 31;
  int qt = blockIdx.x, bh = blockIdx.y;
  int r0 = qt * 16;
  const unsigned short* Q = qh + ((size_t)bh * S_ + r0) * E_;
  ABreg a0, a1;
  loadA_bf(Q, E_, a0);
  loadA_bf(Q + 32, E_, a1);
  int n = lane & 15, hi = (lane >> 4) & 1;
  for (int kt = w; kt <= qt; kt += 8) {
    const unsigned short* K = kh + ((size_t)bh * S_ + kt * 16) * E_;
    f32x8 acc = {};
    { ABreg bb; loadB_bf(K, E_, bb);      acc = wmma_bf16(a0, bb, acc); }
    { ABreg bb; loadB_bf(K + 32, E_, bb); acc = wmma_bf16(a1, bb, acc); }
    int kcol = kt * 16 + n;
#pragma unroll
    for (int c = 0; c < 8; ++c) {
      int qrow = r0 + c + hi * 8;
      float s = acc[c] * 0.125f;                 // 1/sqrt(E=64)
      if (kcol > qrow) s = -__builtin_inff();    // causal mask
      wts[((size_t)bh * S_ + qrow) * S_ + kcol] = s;
    }
  }
}

// ---- K2b: online-softmax rows; fully initializes weights output ------------
__global__ __launch_bounds__(256) void k_softmax(float* __restrict__ wts) {
  __shared__ float redm[16][17];
  __shared__ float reds[16][17];
  int row16 = threadIdx.x >> 4;
  int j = threadIdx.x & 15;
  int q = blockIdx.x * 16 + row16;
  int bh = blockIdx.y;
  float* rowp = wts + ((size_t)bh * S_ + q) * S_;
  int len = q + 1;
  // online max/sum over this thread's strided elements (single read pass)
  float m = -__builtin_inff(), s = 0.f;
  for (int k = j; k < len; k += 16) {
    float x = rowp[k];
    float nm = fmaxf(m, x);
    s = s * __expf(m - nm) + __expf(x - nm);
    m = nm;
  }
  redm[row16][j] = m;
  reds[row16][j] = s;
  __syncthreads();
  float M = -__builtin_inff();
#pragma unroll
  for (int i = 0; i < 16; ++i) M = fmaxf(M, redm[row16][i]);
  float Ssum = 0.f;
#pragma unroll
  for (int i = 0; i < 16; ++i) Ssum += reds[row16][i] * __expf(redm[row16][i] - M);
  float inv = 1.f / Ssum;
  // single write pass over the whole row (zeros above the diagonal)
  for (int k = j; k < S_; k += 16)
    rowp[k] = (k < len) ? __expf(rowp[k] - M) * inv : 0.f;
}

// ---- K2c: attn = P @ V (waves split K, LDS partial reduction) --------------
__global__ __launch_bounds__(256) void k_attnv(const float* __restrict__ wts,           // [BH,S,S]
                                               const unsigned short* __restrict__ vhT,  // [BH,E,S]
                                               unsigned short* __restrict__ attnC) {    // [B,S,D] bf16
  __shared__ float part[8][16][64];   // 32 KB
  int w = threadIdx.x >> 5, lane = threadIdx.x & 31;
  int qt = blockIdx.x, bh = blockIdx.y;
  int r0 = qt * 16;
  int L = (qt + 1) * 16;              // valid K extent (zeros beyond per-row diag)
  int n = lane & 15, hi = (lane >> 4) & 1;
  f32x8 acc[4] = {};
  const float* Prow0 = wts + ((size_t)bh * S_ + r0) * S_;
  const unsigned short* Vb = vhT + (size_t)bh * E_ * S_;
  for (int k0 = w * 32; k0 < L; k0 += 256) {
    ABreg a;                          // A = P tile 16x32 from fp32 (hw cvt_pk)
    const float* row = Prow0 + (size_t)n * S_ + k0 + (hi ? 8 : 0);
#pragma unroll
    for (int i = 0; i < 4; ++i) a.u[i]     = packbf(row[2 * i], row[2 * i + 1]);
#pragma unroll
    for (int i = 0; i < 4; ++i) a.u[4 + i] = packbf(row[16 + 2 * i], row[16 + 2 * i + 1]);
#pragma unroll
    for (int t = 0; t < 4; ++t) {
      ABreg bb; loadB_bf(Vb + (size_t)(t * 16) * S_ + k0, S_, bb);
      acc[t] = wmma_bf16(a, bb, acc[t]);
    }
  }
#pragma unroll
  for (int t = 0; t < 4; ++t)
#pragma unroll
    for (int c = 0; c < 8; ++c)
      part[w][c + hi * 8][t * 16 + n] = acc[t][c];
  __syncthreads();
  int b = bh >> 4, h = bh & 15;
  for (int idx = threadIdx.x; idx < 16 * 64; idx += 256) {
    int m = idx >> 6, e = idx & 63;
    float s = 0.f;
#pragma unroll
    for (int ww = 0; ww < 8; ++ww) s += part[ww][m][e];
    attnC[((size_t)b * S_ + r0 + m) * D_ + h * E_ + e] = f2bf(s);
  }
}

// ---- K3: out = concat @ Wo + bo (B slab async-staged to LDS) ---------------
__global__ __launch_bounds__(256) void k_outproj(const unsigned short* __restrict__ attnC, // [BS,D]
                                                 const unsigned short* __restrict__ WoT,   // [N,D]
                                                 const float* __restrict__ bo,
                                                 float* __restrict__ out) {                // [BS,D]
  __shared__ unsigned short Bs[2][64 * 32];
  int w = threadIdx.x >> 5, lane = threadIdx.x & 31;
  int r0 = blockIdx.x * 128 + w * 16;
  int c0 = blockIdx.y * 64;
  const unsigned short* Wn = WoT + (size_t)c0 * D_;

  f32x8 acc[4] = {};
  stage_B(Wn, D_, Bs[0]);
  int cur = 0;
  for (int d0 = 0; d0 < D_; d0 += 32) {
    stage_wait();
    if (d0 + 32 < D_) stage_B(Wn + d0 + 32, D_, Bs[cur ^ 1]);
    ABreg a; loadA_bf(attnC + (size_t)r0 * D_ + d0, D_, a);
#pragma unroll
    for (int t = 0; t < 4; ++t) {
      ABreg bb; loadB_lds(Bs[cur], t * 16, bb);
      acc[t] = wmma_bf16(a, bb, acc[t]);
    }
    __syncthreads();
    cur ^= 1;
  }

  int n = lane & 15, hi = (lane >> 4) & 1;
#pragma unroll
  for (int t = 0; t < 4; ++t) {
    int col = c0 + t * 16 + n;
    float bvv = bo[col];
#pragma unroll
    for (int c = 0; c < 8; ++c)
      out[((size_t)(r0 + c + hi * 8)) * D_ + col] = acc[t][c] + bvv;
  }
}

extern "C" void kernel_launch(void* const* d_in, const int* in_sizes, int n_in,
                              void* d_out, int out_size, void* d_ws, size_t ws_size,
                              hipStream_t stream) {
  (void)in_sizes; (void)n_in; (void)out_size; (void)ws_size;
  const float* q  = (const float*)d_in[0];
  const float* k  = (const float*)d_in[1];
  const float* v  = (const float*)d_in[2];
  const float* Wq = (const float*)d_in[3];
  const float* bq = (const float*)d_in[4];
  const float* Wk = (const float*)d_in[5];
  const float* bk = (const float*)d_in[6];
  const float* Wv = (const float*)d_in[7];
  const float* bv = (const float*)d_in[8];
  const float* Wo = (const float*)d_in[9];
  const float* bo = (const float*)d_in[10];

  float* out = (float*)d_out;                               // [B,S,D]
  float* wts = (float*)d_out + (size_t)B_ * S_ * D_;        // [B,H,S,S]

  char* ws = (char*)d_ws;
  size_t off = 0;
  auto alloc = [&](size_t bytes) {
    void* p = ws + off;
    off += (bytes + 255) & ~(size_t)255;
    return p;
  };
  unsigned short* qb    = (unsigned short*)alloc((size_t)B_ * S_ * D_ * 2);
  unsigned short* kb    = (unsigned short*)alloc((size_t)B_ * S_ * D_ * 2);
  unsigned short* vb    = (unsigned short*)alloc((size_t)B_ * S_ * D_ * 2);
  unsigned short* WqT   = (unsigned short*)alloc((size_t)H_ * E_ * D_ * 2);
  unsigned short* WkT   = (unsigned short*)alloc((size_t)H_ * E_ * D_ * 2);
  unsigned short* WvT   = (unsigned short*)alloc((size_t)H_ * E_ * D_ * 2);
  unsigned short* WoT   = (unsigned short*)alloc((size_t)D_ * D_ * 2);
  unsigned short* qh    = (unsigned short*)alloc((size_t)BH_ * S_ * E_ * 2);
  unsigned short* kh    = (unsigned short*)alloc((size_t)BH_ * S_ * E_ * 2);
  unsigned short* vhT   = (unsigned short*)alloc((size_t)BH_ * E_ * S_ * 2);
  unsigned short* attnC = (unsigned short*)alloc((size_t)BS_ * D_ * 2);

  dim3 blk(256);
  int ncvt = (B_ * S_ * D_) / 4 / 256;   // float4 per thread
  k_cvt<<<ncvt, blk, 0, stream>>>((const float4*)q, (uint2*)qb);
  k_cvt<<<ncvt, blk, 0, stream>>>((const float4*)k, (uint2*)kb);
  k_cvt<<<ncvt, blk, 0, stream>>>((const float4*)v, (uint2*)vb);

  k_transposeW <<<(H_ * D_ * E_) / 256, blk, 0, stream>>>(Wq, WqT);
  k_transposeW <<<(H_ * D_ * E_) / 256, blk, 0, stream>>>(Wk, WkT);
  k_transposeW <<<(H_ * D_ * E_) / 256, blk, 0, stream>>>(Wv, WvT);
  k_transposeWo<<<(D_ * D_) / 256,      blk, 0, stream>>>(Wo, WoT);

  dim3 g1(S_ / 128, BH_);
  k_qkv_proj<<<g1, blk, 0, stream>>>(qb, WqT, bq, qh, 0);
  k_qkv_proj<<<g1, blk, 0, stream>>>(kb, WkT, bk, kh, 0);
  k_qkv_proj<<<g1, blk, 0, stream>>>(vb, WvT, bv, vhT, 1);

  dim3 g2(S_ / 16, BH_);
  k_scores <<<g2, blk, 0, stream>>>(qh, kh, wts);
  k_softmax<<<g2, blk, 0, stream>>>(wts);
  k_attnv  <<<g2, blk, 0, stream>>>(wts, vhT, attnC);

  dim3 g3(BS_ / 128, D_ / 64);
  k_outproj<<<g3, blk, 0, stream>>>(attnC, WoT, bo, out);
}
